// OrbitalCrystalGraphConvNet_27633819583187
// MI455X (gfx1250) — matrix operand order repaired and checked by reading, not compiled, not run on hardware
//
#include <hip/hip_runtime.h>

typedef __attribute__((ext_vector_type(16))) _Float16 v16h;
typedef __attribute__((ext_vector_type(8)))  _Float16 v8h;
typedef __attribute__((ext_vector_type(8)))  float    v8f;

// Problem constants (from setup_inputs)
constexpr int N_ATOM = 2000;
constexpr int M_NBR  = 12;
constexpr int NM     = N_ATOM * M_NBR;   // 24000
constexpr int OF     = 92;
constexpr int OFP    = 96;               // pad to 3*32
constexpr int A_F    = 64;
constexpr int B_F    = 41;
constexpr int TWOA   = 128;
constexpr int TWOB   = 82;
constexpr int TWOBP  = 96;               // pad output cols to 6*16
constexpr int KTOT   = 2 * A_F + B_F;    // 169
constexpr int KP     = 192;              // pad to 6*32
constexpr int H_F    = 128;
constexpr int NCRY   = 64;
constexpr float BN_EPS = 1e-5f;

__device__ __forceinline__ float sp_f(float x) {
  // numerically stable softplus = log1p(exp(x))
  return x > 0.f ? x + log1pf(__expf(-x)) : log1pf(__expf(x));
}
__device__ __forceinline__ float sg_f(float x) {
  return 1.f / (1.f + __expf(-x));
}

// ---------------------------------------------------------------------------
// WMMA GEMM: C[r, o] = act( sum_k Xh[r,k] * Wh[o,k] + bias[o] )
// Xh: (R, K) f16 row-major, R % (16*RT) == 0, K % 32 == 0
// Wh: (Opad, K) f16 row-major, Opad % (16*OT) == 0
// One wave (32 threads) computes an RT x OT grid of 16x16 tiles: A fragments
// are reused across OT column tiles, B fragments across RT row tiles
// (RT*OT independent accumulators hide WMMA latency).
// A-frag (16x32 f16): lane<16 -> row M=lane, K {0..7,16..23}; lane>=16 -> row
// M=lane-16, K {8..15,24..31}.  B-frag (32x16): lane<16 -> col N=lane, K
// {0..7,16..23}; lane>=16 -> col N=lane-16, K {8..15,24..31} (same k order as A).
// ---------------------------------------------------------------------------
template <int RT, int OT>
__global__ void gemm_xwt(const _Float16* __restrict__ Xh,
                         const _Float16* __restrict__ Wh,
                         const float* __restrict__ bias,
                         float* __restrict__ C,
                         int K, int Ovalid, int ldc, int act)
{
  const int lane = threadIdx.x;
  const int r0 = blockIdx.x * (16 * RT);
  const int o0 = blockIdx.y * (16 * OT);
  const int l  = lane & 15;
  const int hi = lane >> 4;   // 0 or 1

  const _Float16* arow[RT];
#pragma unroll
  for (int rt = 0; rt < RT; ++rt)
    arow[rt] = Xh + (size_t)(r0 + rt * 16 + l) * K;
  const _Float16* brow[OT];
#pragma unroll
  for (int ot = 0; ot < OT; ++ot)
    brow[ot] = Wh + (size_t)(o0 + ot * 16 + l) * K;

  v8f acc[RT][OT];
#pragma unroll
  for (int rt = 0; rt < RT; ++rt)
#pragma unroll
    for (int ot = 0; ot < OT; ++ot)
      acc[rt][ot] = v8f{};

  for (int k0 = 0; k0 < K; k0 += 32) {
    v16h aF[RT];
#pragma unroll
    for (int rt = 0; rt < RT; ++rt) {
      v8h alo = *(const v8h*)(arow[rt] + k0 + hi * 8);
      v8h ahi = *(const v8h*)(arow[rt] + k0 + 16 + hi * 8);
#pragma unroll
      for (int i = 0; i < 8; ++i) { aF[rt][i] = alo[i]; aF[rt][8 + i] = ahi[i]; }
    }
#pragma unroll
    for (int ot = 0; ot < OT; ++ot) {
      v8h blo = *(const v8h*)(brow[ot] + k0 + hi * 8);
      v8h bhi = *(const v8h*)(brow[ot] + k0 + 16 + hi * 8);
      v16h bF;
#pragma unroll
      for (int i = 0; i < 8; ++i) { bF[i] = blo[i]; bF[8 + i] = bhi[i]; }
#pragma unroll
      for (int rt = 0; rt < RT; ++rt)
        acc[rt][ot] = __builtin_amdgcn_wmma_f32_16x16x32_f16(
            false, aF[rt], false, bF, (short)0, acc[rt][ot], false, false);
    }
  }

#pragma unroll
  for (int ot = 0; ot < OT; ++ot) {
    const int col = o0 + ot * 16 + l;
    if (col < Ovalid) {
      float bv = bias ? bias[col] : 0.f;
#pragma unroll
      for (int rt = 0; rt < RT; ++rt) {
#pragma unroll
        for (int j = 0; j < 8; ++j) {
          float v = acc[rt][ot][j] + bv;
          if (act == 1) v = sp_f(v);
          C[(size_t)(r0 + rt * 16 + j + hi * 8) * ldc + col] = v;
        }
      }
    }
  }
}

// ---------------------------------------------------------------------------
// tot[(n*M+m), :] = [x[n] (64) | x[idx[n,m]] (64) | e[n,m] (41) | 0 pad] as f16
// ---------------------------------------------------------------------------
__global__ void build_tot(const float* __restrict__ x, const float* __restrict__ e,
                          const int* __restrict__ idx, _Float16* __restrict__ tot)
{
  const int row = blockIdx.x;          // n*M + m
  const int k   = threadIdx.x;         // 0..191
  const int n   = row / M_NBR;
  float v = 0.f;
  if (k < A_F)            v = x[n * A_F + k];
  else if (k < 2 * A_F) { int j = idx[row]; v = x[j * A_F + (k - A_F)]; }
  else if (k < KTOT)      v = e[(size_t)row * B_F + (k - 2 * A_F)];
  tot[(size_t)row * KP + k] = (_Float16)v;
}

// f32 -> padded f16 (optionally softplus-activated)
__global__ void pack_x(const float* __restrict__ src, _Float16* __restrict__ dst,
                       int R, int Ksrc, int Kpad, int apply_sp)
{
  int tid = blockIdx.x * blockDim.x + threadIdx.x;
  if (tid >= R * Kpad) return;
  int r = tid / Kpad, k = tid - r * Kpad;
  float v = 0.f;
  if (k < Ksrc) { v = src[(size_t)r * Ksrc + k]; if (apply_sp) v = sp_f(v); }
  dst[tid] = (_Float16)v;
}

// copy a f32 weight sub-block into a (pre-zeroed) padded f16 weight buffer
__global__ void pack_block(_Float16* __restrict__ dst, int dstLd, int dstColOff,
                           const float* __restrict__ src, int srcLd, int srcColOff,
                           int rows, int cols)
{
  int tid = blockIdx.x * blockDim.x + threadIdx.x;
  if (tid >= rows * cols) return;
  int r = tid / cols, c = tid - r * cols;
  dst[(size_t)r * dstLd + dstColOff + c] =
      (_Float16)src[(size_t)r * srcLd + srcColOff + c];
}

// ---------------------------------------------------------------------------
// Per-column BN stats (biased var) folded into scale/shift: one block per col
// ---------------------------------------------------------------------------
__global__ void colstats(const float* __restrict__ X, int rows, int ld,
                         const float* __restrict__ g, const float* __restrict__ b,
                         float* __restrict__ scale, float* __restrict__ shift)
{
  const int c = blockIdx.x;
  float s = 0.f, s2 = 0.f;
  for (int r = threadIdx.x; r < rows; r += blockDim.x) {
    float v = X[(size_t)r * ld + c];
    s += v; s2 += v * v;
  }
  __shared__ float sh1[256], sh2[256];
  sh1[threadIdx.x] = s; sh2[threadIdx.x] = s2; __syncthreads();
  for (int o = blockDim.x >> 1; o > 0; o >>= 1) {
    if (threadIdx.x < (unsigned)o) {
      sh1[threadIdx.x] += sh1[threadIdx.x + o];
      sh2[threadIdx.x] += sh2[threadIdx.x + o];
    }
    __syncthreads();
  }
  if (threadIdx.x == 0) {
    float mean = sh1[0] / rows;
    float var  = sh2[0] / rows - mean * mean;
    float sc   = g[c] * rsqrtf(var + BN_EPS);
    scale[c] = sc;
    shift[c] = b[c] - mean * sc;
  }
}

// ---------------------------------------------------------------------------
// BN stats for the virtual (N*M*M, 128) batch h3 = a[n] + u[n,m] + v[n,l].
// Separable: sum  = 144*A + 12*(U+V)
//            sum2 = 144*A^2 + 12*(U2+V2) + 24*A*(U+V) + 2*U*V   per atom n.
// ---------------------------------------------------------------------------
__global__ void stats3(const float* __restrict__ a, const float* __restrict__ u,
                       const float* __restrict__ v,
                       const float* __restrict__ g, const float* __restrict__ b,
                       float* __restrict__ scale, float* __restrict__ shift)
{
  const int c = blockIdx.x;  // 0..127
  float s = 0.f, s2 = 0.f;
  for (int n = threadIdx.x; n < N_ATOM; n += blockDim.x) {
    float A = a[(size_t)n * TWOA + c];
    float U = 0.f, U2 = 0.f, V = 0.f, V2 = 0.f;
#pragma unroll
    for (int m = 0; m < M_NBR; ++m) {
      float uu = u[(size_t)(n * M_NBR + m) * TWOA + c];
      float vv = v[(size_t)(n * M_NBR + m) * TWOA + c];
      U += uu; U2 += uu * uu; V += vv; V2 += vv * vv;
    }
    s  += 144.f * A + 12.f * (U + V);
    s2 += 144.f * A * A + 12.f * (U2 + V2) + 24.f * A * (U + V) + 2.f * U * V;
  }
  __shared__ float sh1[256], sh2[256];
  sh1[threadIdx.x] = s; sh2[threadIdx.x] = s2; __syncthreads();
  for (int o = blockDim.x >> 1; o > 0; o >>= 1) {
    if (threadIdx.x < (unsigned)o) {
      sh1[threadIdx.x] += sh1[threadIdx.x + o];
      sh2[threadIdx.x] += sh2[threadIdx.x + o];
    }
    __syncthreads();
  }
  if (threadIdx.x == 0) {
    const float T = (float)N_ATOM * 144.f;
    float mean = sh1[0] / T;
    float var  = sh2[0] / T - mean * mean;
    float sc   = g[c] * rsqrtf(var + BN_EPS);
    scale[c] = sc;
    shift[c] = b[c] - mean * sc;
  }
}

// two_body[n,c] = sum_m sigmoid(bn(h)[..c]) * softplus(bn(h)[..64+c])
__global__ void two_body(const float* __restrict__ h, const float* __restrict__ s,
                         const float* __restrict__ t, float* __restrict__ tb)
{
  int tid = blockIdx.x * blockDim.x + threadIdx.x;
  if (tid >= N_ATOM * A_F) return;
  int n = tid >> 6, c = tid & 63;
  float s1 = s[c], t1 = t[c], s2 = s[A_F + c], t2 = t[A_F + c];
  float acc = 0.f;
#pragma unroll
  for (int m = 0; m < M_NBR; ++m) {
    const float* hp = h + (size_t)(n * M_NBR + m) * TWOA;
    acc += sg_f(hp[c] * s1 + t1) * sp_f(hp[A_F + c] * s2 + t2);
  }
  tb[tid] = acc;
}

// three_body[n,c] += sum_{m,l} sigmoid/softplus over h3 decomposition
__global__ void three_body(const float* __restrict__ a, const float* __restrict__ u,
                           const float* __restrict__ v, const float* __restrict__ s,
                           const float* __restrict__ t, float* __restrict__ tb)
{
  int tid = blockIdx.x * blockDim.x + threadIdx.x;
  if (tid >= N_ATOM * A_F) return;
  int n = tid >> 6, c = tid & 63;
  float s1 = s[c], s2 = s[A_F + c];
  float base1 = a[(size_t)n * TWOA + c] * s1 + t[c];
  float base2 = a[(size_t)n * TWOA + A_F + c] * s2 + t[A_F + c];
  float p1[M_NBR], p2[M_NBR], q1[M_NBR], q2[M_NBR];
#pragma unroll
  for (int m = 0; m < M_NBR; ++m) {
    const float* up = u + (size_t)(n * M_NBR + m) * TWOA;
    const float* vp = v + (size_t)(n * M_NBR + m) * TWOA;
    p1[m] = up[c] * s1; p2[m] = up[A_F + c] * s2;
    q1[m] = vp[c] * s1; q2[m] = vp[A_F + c] * s2;
  }
  float acc = 0.f;
  for (int m = 0; m < M_NBR; ++m) {
    float b1 = base1 + p1[m], b2 = base2 + p2[m];
#pragma unroll
    for (int l = 0; l < M_NBR; ++l)
      acc += sg_f(b1 + q1[l]) * sp_f(b2 + q2[l]);
  }
  tb[tid] += acc;
}

// e_out = e_in + sigmoid(bn(he)[:B]) * softplus(bn(he)[B:])
__global__ void edge_update(const float* __restrict__ he, const float* __restrict__ s,
                            const float* __restrict__ t, const float* __restrict__ ein,
                            float* __restrict__ eo)
{
  int tid = blockIdx.x * blockDim.x + threadIdx.x;
  if (tid >= NM * B_F) return;
  int r = tid / B_F, c = tid - r * B_F;
  float h1 = he[(size_t)r * TWOBP + c] * s[c] + t[c];
  float h2 = he[(size_t)r * TWOBP + B_F + c] * s[B_F + c] + t[B_F + c];
  eo[tid] = ein[tid] + sg_f(h1) * sp_f(h2);
}

// x_out = softplus(x + bn2(two_body + three_body))
__global__ void update_x(const float* __restrict__ x, const float* __restrict__ tb,
                         const float* __restrict__ s, const float* __restrict__ t,
                         float* __restrict__ xo)
{
  int tid = blockIdx.x * blockDim.x + threadIdx.x;
  if (tid >= N_ATOM * A_F) return;
  int c = tid & 63;
  xo[tid] = sp_f(x[tid] + tb[tid] * s[c] + t[c]);
}

// segment sums for per-crystal mean pooling
__global__ void pool_seg(const float* __restrict__ y, const int* __restrict__ seg,
                         float* __restrict__ psum, float* __restrict__ cnt)
{
  int tid = blockIdx.x * blockDim.x + threadIdx.x;
  if (tid >= N_ATOM * H_F) return;
  int n = tid >> 7, c = tid & 127;
  int k = seg[n];
  atomicAdd(&psum[(size_t)k * H_F + c], y[tid]);
  if (c == 0) atomicAdd(&cnt[k], 1.0f);
}

__global__ void final_out(const float* __restrict__ psum, const float* __restrict__ cnt,
                          const float* __restrict__ W, const float* __restrict__ bo,
                          float* __restrict__ out)
{
  int k = threadIdx.x;
  if (k >= NCRY) return;
  float inv = 1.f / fmaxf(cnt[k], 1.f);
  float acc = 0.f;
  for (int c = 0; c < H_F; ++c) acc += psum[(size_t)k * H_F + c] * inv * W[c];
  out[k] = acc + bo[0];
}

// ---------------------------------------------------------------------------

extern "C" void kernel_launch(void* const* d_in, const int* in_sizes, int n_in,
                              void* d_out, int out_size, void* d_ws, size_t ws_size,
                              hipStream_t stream)
{
  const float* atom_fea = (const float*)d_in[0];
  const float* nbr_fea  = (const float*)d_in[1];
  const int*   nbr_idx  = (const int*)d_in[2];
  const int*   site_seg = (const int*)d_in[3];
  const float* emb_W    = (const float*)d_in[4];
  const float* emb_b    = (const float*)d_in[5];
  const float* fcW      = (const float*)d_in[6];
  const float* fcb      = (const float*)d_in[7];
  const float* bn1_g    = (const float*)d_in[8];
  const float* bn1_b    = (const float*)d_in[9];
  const float* bn2_g    = (const float*)d_in[10];
  const float* bn2_b    = (const float*)d_in[11];
  const float* eW       = (const float*)d_in[12];
  const float* eb       = (const float*)d_in[13];
  const float* bne_g    = (const float*)d_in[14];
  const float* bne_b    = (const float*)d_in[15];
  const float* W3       = (const float*)d_in[16];
  const float* b3       = (const float*)d_in[17];
  const float* bn3_g    = (const float*)d_in[18];
  const float* bn3_b    = (const float*)d_in[19];
  const float* fc1W     = (const float*)d_in[20];
  const float* fc1b     = (const float*)d_in[21];
  const float* outW     = (const float*)d_in[22];
  const float* outb     = (const float*)d_in[23];
  float* out = (float*)d_out;

  // ---- workspace carving (256B aligned) ----
  char* p = (char*)d_ws;
  auto alloc = [&](size_t bytes) -> void* {
    void* r = (void*)p;
    p += (bytes + 255) & ~(size_t)255;
    return r;
  };
  float*    xA    = (float*)alloc(sizeof(float) * N_ATOM * A_F);
  float*    xB    = (float*)alloc(sizeof(float) * N_ATOM * A_F);
  _Float16* xh    = (_Float16*)alloc(sizeof(_Float16) * N_ATOM * A_F);
  _Float16* atomh = (_Float16*)alloc(sizeof(_Float16) * N_ATOM * OFP);
  _Float16* embWh = (_Float16*)alloc(sizeof(_Float16) * A_F * OFP);
  float*    eA    = (float*)alloc(sizeof(float) * NM * B_F);
  float*    eB    = (float*)alloc(sizeof(float) * NM * B_F);
  _Float16* toth  = (_Float16*)alloc(sizeof(_Float16) * (size_t)NM * KP);
  float*    hbuf  = (float*)alloc(sizeof(float) * (size_t)NM * TWOA);
  float*    hebuf = (float*)alloc(sizeof(float) * (size_t)NM * TWOBP);
  float*    ubuf  = (float*)alloc(sizeof(float) * (size_t)NM * TWOA);
  float*    vbuf  = (float*)alloc(sizeof(float) * (size_t)NM * TWOA);
  float*    abuf  = (float*)alloc(sizeof(float) * N_ATOM * TWOA);
  _Float16* fcWh  = (_Float16*)alloc(sizeof(_Float16) * TWOA * KP);
  _Float16* eWh   = (_Float16*)alloc(sizeof(_Float16) * TWOBP * KP);
  _Float16* uWh   = (_Float16*)alloc(sizeof(_Float16) * TWOA * KP);
  _Float16* vWh   = (_Float16*)alloc(sizeof(_Float16) * TWOA * KP);
  _Float16* aWh   = (_Float16*)alloc(sizeof(_Float16) * TWOA * A_F);
  _Float16* fc1Wh = (_Float16*)alloc(sizeof(_Float16) * H_F * A_F);
  float*    s1 = (float*)alloc(sizeof(float) * TWOA);
  float*    t1 = (float*)alloc(sizeof(float) * TWOA);
  float*    se = (float*)alloc(sizeof(float) * TWOBP);
  float*    te = (float*)alloc(sizeof(float) * TWOBP);
  float*    s3 = (float*)alloc(sizeof(float) * TWOA);
  float*    t3 = (float*)alloc(sizeof(float) * TWOA);
  float*    s2 = (float*)alloc(sizeof(float) * A_F);
  float*    t2 = (float*)alloc(sizeof(float) * A_F);
  float*    tb = (float*)alloc(sizeof(float) * N_ATOM * A_F);
  _Float16* zh = (_Float16*)alloc(sizeof(_Float16) * N_ATOM * A_F);
  float*    y  = (float*)alloc(sizeof(float) * N_ATOM * H_F);
  float* psum  = (float*)alloc(sizeof(float) * NCRY * H_F);
  float* cnt   = (float*)alloc(sizeof(float) * NCRY);
  (void)ws_size; (void)in_sizes; (void)n_in; (void)out_size;

  const int TPB = 256;
  auto cdiv = [](int a, int b) { return (a + b - 1) / b; };

  // ---- embedding: x0 = atom_fea @ emb_W.T + emb_b ----
  pack_x<<<cdiv(N_ATOM * OFP, TPB), TPB, 0, stream>>>(atom_fea, atomh, N_ATOM, OF, OFP, 0);
  hipMemsetAsync(embWh, 0, sizeof(_Float16) * A_F * OFP, stream);
  pack_block<<<cdiv(A_F * OF, TPB), TPB, 0, stream>>>(embWh, OFP, 0, emb_W, OF, 0, A_F, OF);
  gemm_xwt<1, 4><<<dim3(N_ATOM / 16, 1), 32, 0, stream>>>(atomh, embWh, emb_b, xA,
                                                          OFP, A_F, A_F, 0);

  const float* xcur = xA;  float* xnext = xB;
  const float* ecur = nbr_fea;  float* enext = eA;

  for (int i = 0; i < 3; ++i) {
    const float* W3i = W3 + (size_t)i * TWOA * (3 * A_F + 2 * B_F);  // (128, 274)
    // pack inputs
    pack_x<<<cdiv(N_ATOM * A_F, TPB), TPB, 0, stream>>>(xcur, xh, N_ATOM, A_F, A_F, 0);
    build_tot<<<NM, KP, 0, stream>>>(xcur, ecur, nbr_idx, toth);
    // pack weights (zero-padded f16)
    hipMemsetAsync(fcWh, 0, sizeof(_Float16) * TWOA * KP, stream);
    hipMemsetAsync(eWh,  0, sizeof(_Float16) * TWOBP * KP, stream);
    hipMemsetAsync(uWh,  0, sizeof(_Float16) * TWOA * KP, stream);
    hipMemsetAsync(vWh,  0, sizeof(_Float16) * TWOA * KP, stream);
    pack_block<<<cdiv(TWOA * KTOT, TPB), TPB, 0, stream>>>(fcWh, KP, 0,
        fcW + (size_t)i * TWOA * KTOT, KTOT, 0, TWOA, KTOT);
    pack_block<<<cdiv(TWOB * KTOT, TPB), TPB, 0, stream>>>(eWh, KP, 0,
        eW + (size_t)i * TWOB * KTOT, KTOT, 0, TWOB, KTOT);
    // uW = [0_64 | Wj | Wij | 0]  (W3 blocks: Wa 0:64, Wj 64:128, Wl 128:192,
    //                              Wij 192:233, Wil 233:274)
    pack_block<<<cdiv(TWOA * A_F, TPB), TPB, 0, stream>>>(uWh, KP, A_F,
        W3i, 3 * A_F + 2 * B_F, A_F, TWOA, A_F);
    pack_block<<<cdiv(TWOA * B_F, TPB), TPB, 0, stream>>>(uWh, KP, 2 * A_F,
        W3i, 3 * A_F + 2 * B_F, 3 * A_F, TWOA, B_F);
    // vW = [0_64 | Wl | Wil | 0]
    pack_block<<<cdiv(TWOA * A_F, TPB), TPB, 0, stream>>>(vWh, KP, A_F,
        W3i, 3 * A_F + 2 * B_F, 2 * A_F, TWOA, A_F);
    pack_block<<<cdiv(TWOA * B_F, TPB), TPB, 0, stream>>>(vWh, KP, 2 * A_F,
        W3i, 3 * A_F + 2 * B_F, 3 * A_F + B_F, TWOA, B_F);
    // aW = Wa (128, 64)
    pack_block<<<cdiv(TWOA * A_F, TPB), TPB, 0, stream>>>(aWh, A_F, 0,
        W3i, 3 * A_F + 2 * B_F, 0, TWOA, A_F);

    // GEMMs (WMMA, register-blocked 2x4 / 2x3 tiles per wave)
    gemm_xwt<2, 4><<<dim3(NM / 32, 2), 32, 0, stream>>>(toth, fcWh,
        fcb + (size_t)i * TWOA, hbuf, KP, TWOA, TWOA, 0);
    gemm_xwt<2, 3><<<dim3(NM / 32, 2), 32, 0, stream>>>(toth, eWh,
        eb + (size_t)i * TWOB, hebuf, KP, TWOB, TWOBP, 0);
    gemm_xwt<2, 4><<<dim3(NM / 32, 2), 32, 0, stream>>>(toth, uWh,
        nullptr, ubuf, KP, TWOA, TWOA, 0);
    gemm_xwt<2, 4><<<dim3(NM / 32, 2), 32, 0, stream>>>(toth, vWh,
        nullptr, vbuf, KP, TWOA, TWOA, 0);
    gemm_xwt<1, 4><<<dim3(N_ATOM / 16, 2), 32, 0, stream>>>(xh, aWh,
        b3 + (size_t)i * TWOA, abuf, A_F, TWOA, TWOA, 0);

    // BN stats (folded scale/shift)
    colstats<<<TWOA, 256, 0, stream>>>(hbuf, NM, TWOA,
        bn1_g + (size_t)i * TWOA, bn1_b + (size_t)i * TWOA, s1, t1);
    colstats<<<TWOB, 256, 0, stream>>>(hebuf, NM, TWOBP,
        bne_g + (size_t)i * TWOB, bne_b + (size_t)i * TWOB, se, te);
    stats3<<<TWOA, 256, 0, stream>>>(abuf, ubuf, vbuf,
        bn3_g + (size_t)i * TWOA, bn3_b + (size_t)i * TWOA, s3, t3);

    // messages
    two_body<<<cdiv(N_ATOM * A_F, TPB), TPB, 0, stream>>>(hbuf, s1, t1, tb);
    three_body<<<cdiv(N_ATOM * A_F, TPB), TPB, 0, stream>>>(abuf, ubuf, vbuf, s3, t3, tb);
    edge_update<<<cdiv(NM * B_F, TPB), TPB, 0, stream>>>(hebuf, se, te, ecur, enext);

    // bn2 over (2000, 64) then x update
    colstats<<<A_F, 256, 0, stream>>>(tb, N_ATOM, A_F,
        bn2_g + (size_t)i * A_F, bn2_b + (size_t)i * A_F, s2, t2);
    update_x<<<cdiv(N_ATOM * A_F, TPB), TPB, 0, stream>>>(xcur, tb, s2, t2, xnext);

    // swap
    const float* xo = xcur; xcur = xnext; xnext = (float*)xo;
    if (i == 0) { ecur = eA; enext = eB; }
    else { const float* eo = ecur; ecur = enext; enext = (float*)eo; }
  }

  // conv_to_fc1: y = softplus(softplus(x) @ fc1W.T + fc1b)
  pack_x<<<cdiv(N_ATOM * A_F, TPB), TPB, 0, stream>>>(xcur, zh, N_ATOM, A_F, A_F, 1);
  pack_block<<<cdiv(H_F * A_F, TPB), TPB, 0, stream>>>(fc1Wh, A_F, 0, fc1W, A_F, 0, H_F, A_F);
  gemm_xwt<1, 4><<<dim3(N_ATOM / 16, 2), 32, 0, stream>>>(zh, fc1Wh, fc1b, y,
                                                          A_F, H_F, H_F, 1);

  // per-crystal mean pooling + output head
  hipMemsetAsync(psum, 0, sizeof(float) * NCRY * H_F, stream);
  hipMemsetAsync(cnt, 0, sizeof(float) * NCRY, stream);
  pool_seg<<<cdiv(N_ATOM * H_F, TPB), TPB, 0, stream>>>(y, site_seg, psum, cnt);
  final_out<<<1, NCRY, 0, stream>>>(psum, cnt, outW, outb, out);
}
